// CombinedAdaptiveSpanRecurrentAttention_49890340110537
// MI455X (gfx1250) — matrix operand order repaired and checked by reading, not compiled, not run on hardware
//
#include <hip/hip_runtime.h>
#include <hip/hip_bf16.h>

typedef __bf16 v16bf __attribute__((ext_vector_type(16)));
typedef __bf16 v8bf  __attribute__((ext_vector_type(8)));
typedef __bf16 v4bf  __attribute__((ext_vector_type(4)));
typedef float  v8f   __attribute__((ext_vector_type(8)));

static constexpr int kB  = 64;    // batch (= attention sequence axis)
static constexpr int kS  = 1024;  // seq len
static constexpr int kD  = 1024;  // model dim
static constexpr int kH  = 16;    // heads
static constexpr int kC  = 256;   // chunk
static constexpr int kDh = 64;    // head dim

// ---------------------------------------------------------------------------
// helpers
// ---------------------------------------------------------------------------
__device__ __forceinline__ v16bf frag16(const __bf16* p0, const __bf16* p1) {
  v8bf lo = *(const v8bf*)p0;
  v8bf hi = *(const v8bf*)p1;
  return __builtin_shufflevector(lo, hi, 0, 1, 2, 3, 4, 5, 6, 7,
                                 8, 9, 10, 11, 12, 13, 14, 15);
}

__device__ __forceinline__ v8f wmma_bf16(v16bf a, v16bf b, v8f c) {
  return __builtin_amdgcn_wmma_f32_16x16x32_bf16(false, a, false, b,
                                                 (short)0, c, false, false);
}

__device__ __forceinline__ v8f vzero8() {
  v8f z = {0.f, 0.f, 0.f, 0.f, 0.f, 0.f, 0.f, 0.f};
  return z;
}

// Async cache->LDS copy (CDNA5 GLOBAL_LOAD_ASYNC_TO_LDS_B128, ASYNCcnt).
// LDS byte offset = low 32 bits of the flat shared-space address (the LDS
// aperture keeps the allocation offset in addr[31:0] per the ISA).
__device__ __forceinline__ void async_b128(void* lds, const void* gptr) {
  unsigned off = (unsigned)(unsigned long long)lds;
  asm volatile("global_load_async_to_lds_b128 %0, %1, off"
               :
               : "v"(off), "v"(gptr)
               : "memory");
}
__device__ __forceinline__ void wait_async0() {
  asm volatile("s_wait_asynccnt 0" ::: "memory");
}

// ---------------------------------------------------------------------------
// fp32 -> bf16 weight conversion
// ---------------------------------------------------------------------------
__global__ __launch_bounds__(256) void cvt_f32_to_bf16(
    const float* __restrict__ x, __bf16* __restrict__ y, int n) {
  int i = (blockIdx.x * 256 + threadIdx.x) * 4;
  if (i < n) {
    float4 f = *(const float4*)(x + i);
    v4bf h;
    h[0] = (__bf16)f.x; h[1] = (__bf16)f.y;
    h[2] = (__bf16)f.z; h[3] = (__bf16)f.w;
    *(v4bf*)(y + i) = h;
  }
}

// ---------------------------------------------------------------------------
// 128x128-tile bf16 WMMA GEMM:  Y[m][n] = X[m][:] . W[n][:] + bias[n]
//   OUT_MODE 0: Y -> q_ws  bf16 scatter [(s*H+h)*B + b][dh]        (M=B*S)
//   OUT_MODE 1: Y -> kv_ws bf16 scatter [(c*H+h)*B + b][dh]; input rows are
//               remapped (b*S + c) so only key[:, :C, :] is consumed (M=B*C)
//   OUT_MODE 2: Y -> fp32 row-major d_out                          (M=B*S)
// Double-buffered LDS; B tiles (and bf16 A tiles) staged with async-to-LDS.
// ---------------------------------------------------------------------------
template <int OUT_MODE, bool A_BF16>
__global__ __launch_bounds__(256) void gemm128x128(
    const void* __restrict__ Xv, const __bf16* __restrict__ W,
    const float* __restrict__ bias, void* __restrict__ Yv) {
  const int tid  = threadIdx.x;
  const int lane = tid & 31;
  const int wave = tid >> 5;
  const int wm   = wave & 1;   // 2 wave rows  (64 M each)
  const int wn   = wave >> 1;  // 4 wave cols  (32 N each)
  const int m0   = blockIdx.y * 128;
  const int n0   = blockIdx.x * 128;
  const int lr   = lane & 15;
  const bool lo16 = (lane & 16) == 0;

  __shared__ __bf16 sA[2][128 * 40];  // [m][k], k-contiguous, pad 40
  __shared__ __bf16 sB[2][128 * 40];  // [n][k], k-contiguous, pad 40

  const float*  Xf = (const float*)Xv;
  const __bf16* Xb = (const __bf16*)Xv;

  v8f acc[4][2];
#pragma unroll
  for (int i = 0; i < 4; i++)
#pragma unroll
    for (int j = 0; j < 2; j++) acc[i][j] = vzero8();

  float4 areg[4];  // in-flight fp32 A chunks (VGPR path)

  auto issueA = [&](int k0) {  // fp32 A: issue global loads only
    if constexpr (!A_BF16) {
#pragma unroll
      for (int i = 0; i < 4; i++) {
        int chunk = tid + i * 256;  // 1024 float4-chunks of a 128x32 tile
        int row = chunk >> 3, c4 = (chunk & 7) << 2;
        int gm = m0 + row;
        size_t xr = (OUT_MODE == 1) ? ((size_t)(gm >> 8) * kS + (gm & (kC - 1)))
                                    : (size_t)gm;
        areg[i] = *(const float4*)(Xf + xr * kD + k0 + c4);
      }
    }
  };
  auto commitA = [&](int buf) {  // fp32 A: cvt + LDS store
    if constexpr (!A_BF16) {
#pragma unroll
      for (int i = 0; i < 4; i++) {
        int chunk = tid + i * 256;
        int row = chunk >> 3, c4 = (chunk & 7) << 2;
        v4bf h;
        h[0] = (__bf16)areg[i].x; h[1] = (__bf16)areg[i].y;
        h[2] = (__bf16)areg[i].z; h[3] = (__bf16)areg[i].w;
        *(v4bf*)(&sA[buf][row * 40 + c4]) = h;
      }
    }
  };
  auto asyncA = [&](int buf, int k0) {  // bf16 A: async straight to LDS
    if constexpr (A_BF16) {
#pragma unroll
      for (int i = 0; i < 2; i++) {
        int chunk = tid + i * 256;
        int row = chunk >> 2, c8 = (chunk & 3) << 3;
        async_b128(&sA[buf][row * 40 + c8],
                   Xb + (size_t)(m0 + row) * kD + k0 + c8);
      }
    }
  };
  auto stageB = [&](int buf, int k0) {  // weights bf16: async to LDS
#pragma unroll
    for (int i = 0; i < 2; i++) {
      int chunk = tid + i * 256;
      int row = chunk >> 2, c8 = (chunk & 3) << 3;
      async_b128(&sB[buf][row * 40 + c8],
                 W + (size_t)(n0 + row) * kD + k0 + c8);
    }
  };

  // prologue: fill buffer 0
  issueA(0);
  asyncA(0, 0);
  stageB(0, 0);
  commitA(0);
  wait_async0();
  __syncthreads();

  const int koffA = lo16 ? 0 : 8;   // A: lanes hold K chunks {koff, koff+16}
  const int koffB = lo16 ? 0 : 8;   // B: lanes hold K chunks {koff, koff+16}

  for (int it = 0; it < kD / 32; ++it) {
    const int buf = it & 1, nxt = buf ^ 1;
    const int k1 = (it + 1) * 32;
    if (k1 < kD) {  // issue next tile's traffic before computing this one
      issueA(k1);
      asyncA(nxt, k1);
      stageB(nxt, k1);
      if (k1 + 32 < kD)
        __builtin_prefetch(W + (size_t)(n0 + (tid >> 1)) * kD + k1 + 32, 0, 3);
    }

    v16bf afrag[4], bfrag[2];
#pragma unroll
    for (int mi = 0; mi < 4; mi++) {
      const __bf16* p = &sA[buf][(wm * 64 + mi * 16 + lr) * 40 + koffA];
      afrag[mi] = frag16(p, p + 16);
    }
#pragma unroll
    for (int ni = 0; ni < 2; ni++) {
      const __bf16* p = &sB[buf][(wn * 32 + ni * 16 + lr) * 40 + koffB];
      bfrag[ni] = frag16(p, p + 16);
    }
#pragma unroll
    for (int mi = 0; mi < 4; mi++)
#pragma unroll
      for (int ni = 0; ni < 2; ni++)
        acc[mi][ni] = wmma_bf16(afrag[mi], bfrag[ni], acc[mi][ni]);

    if (k1 < kD) commitA(nxt);  // cvt+store next A behind the WMMAs
    wait_async0();
    __syncthreads();
  }

  // ---- epilogue: bias + scatter/store ----
  float bv[2];
  bv[0] = bias[n0 + wn * 32 + lr];
  bv[1] = bias[n0 + wn * 32 + 16 + lr];
#pragma unroll
  for (int mi = 0; mi < 4; mi++) {
#pragma unroll
    for (int ni = 0; ni < 2; ni++) {
#pragma unroll
      for (int r = 0; r < 8; r++) {
        int gm = m0 + wm * 64 + mi * 16 + r + (lo16 ? 0 : 8);
        int gn = n0 + wn * 32 + ni * 16 + lr;
        float val = acc[mi][ni][r] + bv[ni];
        if constexpr (OUT_MODE == 0) {
          int b = gm >> 10, s = gm & (kS - 1);      // gm = b*S + s
          int h = gn >> 6,  t = gn & (kDh - 1);     // gn = h*dh + t
          ((__bf16*)Yv)[(((size_t)(s * kH + h)) * kB + b) * kDh + t] = (__bf16)val;
        } else if constexpr (OUT_MODE == 1) {
          int b = gm >> 8, c = gm & (kC - 1);       // gm = b*C + c
          int h = gn >> 6, t = gn & (kDh - 1);
          ((__bf16*)Yv)[(((size_t)(c * kH + h)) * kB + b) * kDh + t] = (__bf16)val;
        } else {
          ((float*)Yv)[(size_t)gm * kD + gn] = val;
        }
      }
    }
  }
}

// ---------------------------------------------------------------------------
// Attention over the batch axis. One workgroup per group g = (n*C+c)*H + h.
// q group tile: 64x64 bf16 contiguous at qw + g*4096
// k/v group tile: at kw/vw + (g % (C*H))*4096
// S = (Q Kᵀ)/8 -> softmax rows -> O = P V -> ctx[b][s][d] bf16 row-major
// ---------------------------------------------------------------------------
__global__ __launch_bounds__(256) void attn_batch64(
    const __bf16* __restrict__ qw, const __bf16* __restrict__ kw,
    const __bf16* __restrict__ vw, __bf16* __restrict__ ctx) {
  const int g  = blockIdx.x;               // (n*C+c)*H + h
  const int gk = g & (kC * kH - 1);        // c*H + h
  const int tid  = threadIdx.x;
  const int lane = tid & 31;
  const int wave = tid >> 5;
  const int lr   = lane & 15;
  const bool lo16 = (lane & 16) == 0;

  __shared__ __bf16 sQ[64 * 72];   // [qb][t]
  __shared__ __bf16 sK[64 * 72];   // [kb][t]   (K-major for B fragments)
  __shared__ __bf16 sVT[64 * 72];  // [t][kb]   (transposed V, K-major)
  __shared__ float  sS[64 * 66];   // scores fp32
  __shared__ __bf16 sP[64 * 72];   // probs bf16

  {  // stage: Q/K via async-to-LDS; V through VGPRs (needs transpose)
    int row = tid >> 2;            // 0..63
    int c0  = (tid & 3) << 4;      // 0,16,32,48
    const __bf16* qp = qw + (size_t)g * 4096 + row * 64 + c0;
    const __bf16* kp = kw + (size_t)gk * 4096 + row * 64 + c0;
    const __bf16* vp = vw + (size_t)gk * 4096 + row * 64 + c0;
    async_b128(sQ + row * 72 + c0,     qp);
    async_b128(sQ + row * 72 + c0 + 8, qp + 8);
    async_b128(sK + row * 72 + c0,     kp);
    async_b128(sK + row * 72 + c0 + 8, kp + 8);
    v8bf v0 = *(const v8bf*)vp;
    v8bf v1 = *(const v8bf*)(vp + 8);
#pragma unroll
    for (int j = 0; j < 8; j++) {
      sVT[(c0 + j) * 72 + row]     = v0[j];
      sVT[(c0 + 8 + j) * 72 + row] = v1[j];
    }
    wait_async0();
  }
  __syncthreads();

  const int mt  = wave >> 1;         // wave's 16-row strip (qb)
  const int ntb = (wave & 1) << 1;   // 2 col tiles per wave
  const int koffA = lo16 ? 0 : 8;
  const int koffB = lo16 ? 0 : 8;

  // ---- GEMM 1: S = Q Kᵀ  (A rows=qb K=t;  B cols=kb, K=t from sK[kb][t]) ----
  v8f acc[2] = {vzero8(), vzero8()};
#pragma unroll
  for (int ks = 0; ks < 2; ks++) {
    const __bf16* pa = sQ + (mt * 16 + lr) * 72 + ks * 32 + koffA;
    v16bf a = frag16(pa, pa + 16);
#pragma unroll
    for (int ni = 0; ni < 2; ni++) {
      const __bf16* pb = sK + ((ntb + ni) * 16 + lr) * 72 + ks * 32 + koffB;
      acc[ni] = wmma_bf16(a, frag16(pb, pb + 16), acc[ni]);
    }
  }
  const float scale = 0.125f;  // 1/sqrt(dh=64)
#pragma unroll
  for (int ni = 0; ni < 2; ni++)
#pragma unroll
    for (int r = 0; r < 8; r++) {
      int row = mt * 16 + r + (lo16 ? 0 : 8);
      int col = (ntb + ni) * 16 + lr;
      sS[row * 66 + col] = acc[ni][r] * scale;
    }
  __syncthreads();

  // ---- softmax: 4 lanes per row, wave32 shuffles ----
  {
    int row = tid >> 2;
    int c0  = (tid & 3) << 4;
    float m = -3.0e38f;
#pragma unroll
    for (int j = 0; j < 16; j++) m = fmaxf(m, sS[row * 66 + c0 + j]);
    m = fmaxf(m, __shfl_xor(m, 1, 32));
    m = fmaxf(m, __shfl_xor(m, 2, 32));
    float e[16], sum = 0.f;
#pragma unroll
    for (int j = 0; j < 16; j++) {
      e[j] = __expf(sS[row * 66 + c0 + j] - m);
      sum += e[j];
    }
    sum += __shfl_xor(sum, 1, 32);
    sum += __shfl_xor(sum, 2, 32);
    float inv = 1.0f / sum;
#pragma unroll
    for (int j = 0; j < 16; j++)
      sP[row * 72 + c0 + j] = (__bf16)(e[j] * inv);
  }
  __syncthreads();

  // ---- GEMM 2: O = P V  (A=sP[qb][kb]; B cols=t, K=kb from sVT[t][kb]) ----
  v8f oacc[2] = {vzero8(), vzero8()};
#pragma unroll
  for (int ks = 0; ks < 2; ks++) {
    const __bf16* pa = sP + (mt * 16 + lr) * 72 + ks * 32 + koffA;
    v16bf a = frag16(pa, pa + 16);
#pragma unroll
    for (int ni = 0; ni < 2; ni++) {
      const __bf16* pb = sVT + ((ntb + ni) * 16 + lr) * 72 + ks * 32 + koffB;
      oacc[ni] = wmma_bf16(a, frag16(pb, pb + 16), oacc[ni]);
    }
  }
  // ctx[b][s][d] bf16, s = g>>4 (= n*C+c since H=16), d = (g&15)*64 + t
  const int s = g >> 4;
  const int h = g & (kH - 1);
#pragma unroll
  for (int ni = 0; ni < 2; ni++)
#pragma unroll
    for (int r = 0; r < 8; r++) {
      int qb = mt * 16 + r + (lo16 ? 0 : 8);  // attention row = batch index
      int t  = (ntb + ni) * 16 + lr;
      ctx[((size_t)qb * kS + s) * kD + h * kDh + t] = (__bf16)oacc[ni][r];
    }
}

// ---------------------------------------------------------------------------
extern "C" void kernel_launch(void* const* d_in, const int* in_sizes, int n_in,
                              void* d_out, int out_size, void* d_ws,
                              size_t ws_size, hipStream_t stream) {
  const float* query     = (const float*)d_in[0];
  const float* key       = (const float*)d_in[1];
  const float* value     = (const float*)d_in[2];
  const float* in_proj_w = (const float*)d_in[3];
  const float* in_proj_b = (const float*)d_in[4];
  const float* out_w     = (const float*)d_in[5];
  const float* out_b     = (const float*)d_in[6];

  const size_t MB = 1024ull * 1024ull;
  char* ws = (char*)d_ws;
  __bf16* wproj = (__bf16*)(ws);               // 3M bf16 (6 MB)
  __bf16* wout  = (__bf16*)(ws + 6 * MB);      // 1M bf16 (2 MB)
  __bf16* q_ws  = (__bf16*)(ws + 8 * MB);      // 64M  bf16 (128 MB)
  __bf16* k_ws  = (__bf16*)(ws + 136 * MB);    // 16M  bf16 (32 MB)
  __bf16* v_ws  = (__bf16*)(ws + 168 * MB);    // 16M  bf16 (32 MB)
  __bf16* c_ws  = (__bf16*)(ws + 200 * MB);    // 64M  bf16 (128 MB)

  const dim3 blk(256);

  // weights -> bf16
  cvt_f32_to_bf16<<<3 * 1024, blk, 0, stream>>>(in_proj_w, wproj, 3 * kD * kD);
  cvt_f32_to_bf16<<<1024, blk, 0, stream>>>(out_w, wout, kD * kD);

  // q projection: (B*S) x D x D, scatter to attention layout
  gemm128x128<0, false><<<dim3(8, 512), blk, 0, stream>>>(
      query, wproj, in_proj_b, q_ws);
  // k / v projections on key[:, :C, :] / value[:, :C, :]: (B*C) x D x D
  gemm128x128<1, false><<<dim3(8, 128), blk, 0, stream>>>(
      key, wproj + (size_t)kD * kD, in_proj_b + kD, k_ws);
  gemm128x128<1, false><<<dim3(8, 128), blk, 0, stream>>>(
      value, wproj + 2 * (size_t)kD * kD, in_proj_b + 2 * kD, v_ws);

  // 16384 independent 64-wide attention groups
  attn_batch64<<<kS * kH, blk, 0, stream>>>(q_ws, k_ws, v_ws, c_ws);

  // output projection: (B*S) x D x D -> fp32 d_out
  gemm128x128<2, true><<<dim3(8, 512), blk, 0, stream>>>(
      c_ws, wout, out_b, d_out);
}
